// NeuralODE_85547158601965
// MI455X (gfx1250) — matrix operand order repaired and checked by reading, not compiled, not run on hardware
//
#include <hip/hip_runtime.h>

typedef __attribute__((ext_vector_type(16))) __bf16 v16bf;
typedef __attribute__((ext_vector_type(8)))  __bf16 bf16x8;
typedef __attribute__((ext_vector_type(8)))  float  v8f;

constexpr int BATCH  = 8192;
constexpr int DIM    = 128;
constexpr int HID    = 512;
constexpr int NSTEPS = 64;

// ---------------------------------------------------------------------------
// Weight transpose + fp32 -> bf16 convert.  dst[n*K + k] = src[k*N + n]
// Produces W^T in out-feature-major layout: the WMMA A operand.
// ---------------------------------------------------------------------------
__global__ __launch_bounds__(256) void transpose_bf16_kernel(
    const float* __restrict__ src, __bf16* __restrict__ dst, int K, int N) {
  int idx = blockIdx.x * blockDim.x + threadIdx.x;
  if (idx >= K * N) return;
  int n = idx / K;
  int k = idx - n * K;
  dst[idx] = (__bf16)src[k * N + n];
}

__global__ __launch_bounds__(256) void copy_f32_kernel(
    const float* __restrict__ src, float* __restrict__ dst, int n) {
  int i = blockIdx.x * blockDim.x + threadIdx.x;
  if (i < n) dst[i] = src[i];
}

// ---------------------------------------------------------------------------
// Stage combination: out = y + dt * (c1*k1 + ... + c6*k6); bf16 out feeds the
// next GEMM's B operand, fp32 out is the in-place 5th-order state update.
// dt computed on-device from ts[] (device mem is unreadable on host in capture).
// ---------------------------------------------------------------------------
template <typename OutT>
__global__ __launch_bounds__(256) void combine_kernel(
    OutT* __restrict__ out, const float* __restrict__ y,
    const float* __restrict__ ts,
    const float* __restrict__ k1, const float* __restrict__ k2,
    const float* __restrict__ k3, const float* __restrict__ k4,
    const float* __restrict__ k5, const float* __restrict__ k6,
    float c1, float c2, float c3, float c4, float c5, float c6, int nk) {
  int i = blockIdx.x * blockDim.x + threadIdx.x;
  if (i >= BATCH * DIM) return;
  float dt  = (ts[1] - ts[0]) * (1.0f / NSTEPS);
  float acc = 0.0f;
  if (nk > 0) acc += c1 * k1[i];
  if (nk > 1) acc += c2 * k2[i];
  if (nk > 2) acc += c3 * k3[i];
  if (nk > 3) acc += c4 * k4[i];
  if (nk > 4) acc += c5 * k5[i];
  if (nk > 5) acc += c6 * k6[i];
  out[i] = (OutT)(y[i] + dt * acc);
}

// ---------------------------------------------------------------------------
// WMMA GEMM, transposed formulation:  D = W^T (A op) x X^T (B op)
//   => out[b][f] = sum_k X[b][k] * W[k][f] + bias[f]   (Y row-major)
//
//   WT : bf16, [NOUT][K]  (WT[f*K + k] == W[k][f])  -> A fragments
//   X  : bf16, [BATCH][K] row-major                  -> B fragments (contiguous)
//   out: [BATCH][NOUT] row-major, 8 contiguous features per lane store.
//
// Each wave: 32 out-features (2 M-tiles) x 64 batch (4 N-tiles),
// K in 32-chunks via v_wmma_f32_16x16x32_bf16, f32 accumulate.
// Fragments are explicitly double-buffered: loads for K-block i+1 are issued
// before the 8 WMMAs of block i, so VMEM latency overlaps the matrix pipe.
//
// Fragment layouts used here (wave32; contiguous k order, not the ISA table order):
//  A 16x32 bf16 : lane l -> row m = l&15, khalf = l>>4;
//                 elem e = K[khalf*16+e] (same order as B)
//  B 32x16 bf16 : lane l -> col n = l&15, khalf = l>>4; elem e = K[khalf*16+e]
//  C/D 16x16 f32: lane l -> col n = l&15; vgpr r -> row m = (l>>4)*8 + r
// ---------------------------------------------------------------------------
union AFrag {
  v16bf  v;
  bf16x8 h[2];
};

template <int K, int NOUT, bool TANH, typename OutT>
__global__ __launch_bounds__(256) void gemm_kernel(
    const __bf16* __restrict__ WT, const __bf16* __restrict__ X,
    const float* __restrict__ bias, OutT* __restrict__ out) {
  constexpr int MS = NOUT / 32;  // feature strips (fast index: WG shares batch strip)
  constexpr int NB = K / 32;     // K blocks
  const int lane   = threadIdx.x & 31;
  const int wave   = blockIdx.x * (blockDim.x >> 5) + (threadIdx.x >> 5);
  const int f0w    = (wave % MS) * 32;   // this wave's first out-feature
  const int b0     = (wave / MS) * 64;   // this wave's first batch row
  const int mrow   = lane & 15;
  const int khalf  = lane >> 4;

  v8f acc[2][4] = {};

  // A operand: transposed weights (out-feature rows)
  const __bf16* a0p = WT + (f0w + mrow) * K + khalf * 16;
  const __bf16* a1p = WT + (f0w + 16 + mrow) * K + khalf * 16;
  // B operand: activations, natural row-major (batch columns)
  const __bf16* bp[4] = {
      X + (b0 + 0 * 16 + mrow) * K + khalf * 16,
      X + (b0 + 1 * 16 + mrow) * K + khalf * 16,
      X + (b0 + 2 * 16 + mrow) * K + khalf * 16,
      X + (b0 + 3 * 16 + mrow) * K + khalf * 16};

  // double-buffered fragment registers
  AFrag a0[2], a1[2];
  v16bf bf[2][4];

  auto load_block = [&](int kb, int buf) {
    a0[buf].h[0] = *(const bf16x8*)(a0p + kb);
    a0[buf].h[1] = *(const bf16x8*)(a0p + kb + 8);
    a1[buf].h[0] = *(const bf16x8*)(a1p + kb);
    a1[buf].h[1] = *(const bf16x8*)(a1p + kb + 8);
#pragma unroll
    for (int nt = 0; nt < 4; ++nt) bf[buf][nt] = *(const v16bf*)(bp[nt] + kb);
  };

  load_block(0, 0);
#pragma unroll
  for (int i = 0; i < NB; ++i) {
    const int cur = i & 1;
    if (i + 1 < NB) load_block((i + 1) * 32, cur ^ 1);  // prefetch next block
#pragma unroll
    for (int nt = 0; nt < 4; ++nt) {
      acc[0][nt] = __builtin_amdgcn_wmma_f32_16x16x32_bf16(
          false, a0[cur].v, false, bf[cur][nt], (short)0, acc[0][nt], false, false);
      acc[1][nt] = __builtin_amdgcn_wmma_f32_16x16x32_bf16(
          false, a1[cur].v, false, bf[cur][nt], (short)0, acc[1][nt], false, false);
    }
  }

  // Epilogue: lane owns batch row b, 8 contiguous out-features -> vector stores.
  const int rbase = khalf * 8;
#pragma unroll
  for (int nt = 0; nt < 4; ++nt) {
    const int b = b0 + nt * 16 + mrow;
#pragma unroll
    for (int mt = 0; mt < 2; ++mt) {
      const int f0 = f0w + mt * 16 + rbase;
      const v8f bv = *(const v8f*)(bias + f0);
      if constexpr (sizeof(OutT) == 2) {
        bf16x8 o;
#pragma unroll
        for (int r = 0; r < 8; ++r) {
          float v = acc[mt][nt][r] + bv[r];
          if (TANH) v = tanhf(v);
          o[r] = (__bf16)v;
        }
        *(bf16x8*)((__bf16*)out + b * NOUT + f0) = o;   // one b128 store
      } else {
        v8f o;
#pragma unroll
        for (int r = 0; r < 8; ++r) {
          float v = acc[mt][nt][r] + bv[r];
          if (TANH) v = tanhf(v);
          o[r] = v;
        }
        *(v8f*)((float*)out + b * NOUT + f0) = o;       // two b128 stores
      }
    }
  }
}

// ---------------------------------------------------------------------------
// Host-side orchestration
// ---------------------------------------------------------------------------
extern "C" void kernel_launch(void* const* d_in, const int* in_sizes, int n_in,
                              void* d_out, int out_size, void* d_ws, size_t ws_size,
                              hipStream_t stream) {
  (void)in_sizes; (void)n_in; (void)out_size; (void)ws_size;

  const float* y0 = (const float*)d_in[0];
  const float* ts = (const float*)d_in[1];
  const float* W1 = (const float*)d_in[2];
  const float* b1 = (const float*)d_in[3];
  const float* W2 = (const float*)d_in[4];
  const float* b2 = (const float*)d_in[5];
  const float* W3 = (const float*)d_in[6];
  const float* b3 = (const float*)d_in[7];
  float* yT = (float*)d_out;

  // workspace carve-up
  char* p = (char*)d_ws;
  __bf16* W1T  = (__bf16*)p; p += (size_t)DIM * HID * 2;   // [HID][DIM]
  __bf16* W2T  = (__bf16*)p; p += (size_t)HID * HID * 2;   // [HID][HID]
  __bf16* W3T  = (__bf16*)p; p += (size_t)HID * DIM * 2;   // [DIM][HID]
  float*  yst  = (float*)p;  p += (size_t)BATCH * DIM * 4; // integrator state
  __bf16* ytmp = (__bf16*)p; p += (size_t)BATCH * DIM * 2; // GEMM-1 B operand
  __bf16* h1   = (__bf16*)p; p += (size_t)BATCH * HID * 2;
  __bf16* h2   = (__bf16*)p; p += (size_t)BATCH * HID * 2;
  float*  kb[6];
  for (int i = 0; i < 6; ++i) { kb[i] = (float*)p; p += (size_t)BATCH * DIM * 4; }

  // --- per-call prep: weights -> bf16 W^T, state <- y0 ---
  transpose_bf16_kernel<<<(DIM * HID + 255) / 256, 256, 0, stream>>>(W1, W1T, DIM, HID);
  transpose_bf16_kernel<<<(HID * HID + 255) / 256, 256, 0, stream>>>(W2, W2T, HID, HID);
  transpose_bf16_kernel<<<(HID * DIM + 255) / 256, 256, 0, stream>>>(W3, W3T, HID, DIM);
  copy_f32_kernel<<<(BATCH * DIM + 255) / 256, 256, 0, stream>>>(y0, yst, BATCH * DIM);

  const int combGrid = (BATCH * DIM + 255) / 256;
  const int gridWide = (HID / 32) * (BATCH / 64) / 8;  // 256 WGs, layers 1&2
  const int gridNarr = (DIM / 32) * (BATCH / 64) / 8;  // 64 WGs, layer 3

  auto mlp = [&](float* kout) {
    gemm_kernel<DIM, HID, true,  __bf16><<<gridWide, 256, 0, stream>>>(W1T, ytmp, b1, h1);
    gemm_kernel<HID, HID, true,  __bf16><<<gridWide, 256, 0, stream>>>(W2T, h1,   b2, h2);
    gemm_kernel<HID, DIM, false, float ><<<gridNarr, 256, 0, stream>>>(W3T, h2,   b3, kout);
  };
  auto combine_bf16 = [&](int nk, const float* c) {
    combine_kernel<__bf16><<<combGrid, 256, 0, stream>>>(
        ytmp, yst, ts, kb[0], kb[1], kb[2], kb[3], kb[4], kb[5],
        c[0], c[1], c[2], c[3], c[4], c[5], nk);
  };

  // Tsit5 tableau
  static const float CZ[6] = {0, 0, 0, 0, 0, 0};
  static const float C2[6] = {0.161f, 0, 0, 0, 0, 0};
  static const float C3[6] = {-0.008480655492356989f, 0.335480655492357f, 0, 0, 0, 0};
  static const float C4[6] = {2.8971530571054935f, -6.359448489975075f, 4.3622954328695815f, 0, 0, 0};
  static const float C5[6] = {5.325864828439257f, -11.748883564062828f, 7.4955393428898365f,
                              -0.09249506636175525f, 0, 0};
  static const float C6[6] = {5.86145544294642f, -12.92096931784711f, 8.159367898576159f,
                              -0.071584973281401f, -0.028269050394068383f, 0};
  static const float CB[6] = {0.09646076681806523f, 0.01f, 0.4798896504144996f,
                              1.379008574103742f, -3.290069515436081f, 2.324710524099774f};
  const float* SC[6] = {CZ, C2, C3, C4, C5, C6};

  for (int step = 0; step < NSTEPS; ++step) {
    for (int s = 0; s < 6; ++s) {
      combine_bf16(s, SC[s]);   // ytmp = bf16(y + dt * sum a_sj k_j)
      mlp(kb[s]);               // k_{s+1} = MLP(ytmp)
    }
    // y <- y + dt * sum b_i k_i   (in place, elementwise)
    combine_kernel<float><<<combGrid, 256, 0, stream>>>(
        yst, yst, ts, kb[0], kb[1], kb[2], kb[3], kb[4], kb[5],
        CB[0], CB[1], CB[2], CB[3], CB[4], CB[5], 6);
  }

  copy_f32_kernel<<<(BATCH * DIM + 255) / 256, 256, 0, stream>>>(yst, yT, BATCH * DIM);
}
